// PointNetPPFrameClassifier_86268713107550
// MI455X (gfx1250) — matrix operand order repaired and checked by reading, not compiled, not run on hardware
//
#include <hip/hip_runtime.h>
#include <hip/hip_bf16.h>

typedef _Float16 v8h  __attribute__((ext_vector_type(8)));
typedef _Float16 v16h __attribute__((ext_vector_type(16)));
typedef float    v8f  __attribute__((ext_vector_type(8)));

// ---------------------------------------------------------------------------
// Farthest point sampling: one block per frame, replicates jnp.argmax
// (first-max) tie semantics via (dist desc, index asc) reduction.
// ---------------------------------------------------------------------------
__global__ void fps_kernel(const float* __restrict__ pos, int* __restrict__ idx,
                           int N, int n) {
  const int f = blockIdx.x;
  const float* p = pos + (size_t)f * N * 3;
  int* out = idx + (size_t)f * n;
  __shared__ float sd[2048];
  __shared__ float rb[256];
  __shared__ int   ri[256];
  __shared__ float cur[3];
  const int tid = threadIdx.x;
  if (tid == 0) { out[0] = 0; cur[0] = p[0]; cur[1] = p[1]; cur[2] = p[2]; }
  __syncthreads();
  for (int j = tid; j < N; j += 256) {
    float dx = p[3*j] - cur[0], dy = p[3*j+1] - cur[1], dz = p[3*j+2] - cur[2];
    sd[j] = dx*dx + dy*dy + dz*dz;
  }
  __syncthreads();
  for (int i = 1; i < n; ++i) {
    float bd = -1.0f; int bi = 0x7fffffff;
    for (int j = tid; j < N; j += 256) {
      float d = sd[j];
      if (d > bd || (d == bd && j < bi)) { bd = d; bi = j; }
    }
    rb[tid] = bd; ri[tid] = bi;
    __syncthreads();
    for (int s = 128; s > 0; s >>= 1) {
      if (tid < s) {
        float od = rb[tid+s]; int oi = ri[tid+s];
        if (od > rb[tid] || (od == rb[tid] && oi < ri[tid])) { rb[tid] = od; ri[tid] = oi; }
      }
      __syncthreads();
    }
    const int nxt = ri[0];
    if (tid == 0) {
      out[i] = nxt;
      cur[0] = p[3*nxt]; cur[1] = p[3*nxt+1]; cur[2] = p[3*nxt+2];
    }
    __syncthreads();
    for (int j = tid; j < N; j += 256) {
      float dx = p[3*j] - cur[0], dy = p[3*j+1] - cur[1], dz = p[3*j+2] - cur[2];
      float d = dx*dx + dy*dy + dz*dz;
      if (d < sd[j]) sd[j] = d;
    }
    __syncthreads();
  }
}

__global__ void gather_pos_kernel(const float* __restrict__ pos, const int* __restrict__ idx,
                                  float* __restrict__ qpos, int N, int n, int total) {
  int t = blockIdx.x * 256 + threadIdx.x;
  if (t >= total) return;
  int f = t / n, i = t - f * n;
  int src = idx[(size_t)f * n + i];
  const float* s = pos + ((size_t)f * N + src) * 3;
  float* d = qpos + ((size_t)f * n + i) * 3;
  d[0] = s[0]; d[1] = s[1]; d[2] = s[2];
}

// ---------------------------------------------------------------------------
// Radius-KNN: one thread per query, 64-entry max-heap ordered by (d2, idx)
// lexicographic -> matches lax.top_k(-d2) tie-breaking. -1 marks invalid.
// ---------------------------------------------------------------------------
__global__ void knn_kernel(const float* __restrict__ pos, const float* __restrict__ qpos,
                           int* __restrict__ nidx, int N, int M, int total, float r2) {
  int t = blockIdx.x * 256 + threadIdx.x;
  if (t >= total) return;
  int f = t / M, q = t - f * M;
  const float* p  = pos  + (size_t)f * N * 3;
  const float* qp = qpos + ((size_t)f * M + q) * 3;
  float qx = qp[0], qy = qp[1], qz = qp[2];
  float hd[64]; int hi[64];
  int cnt = 0;
  for (int j = 0; j < N; ++j) {
    float dx = p[3*j] - qx, dy = p[3*j+1] - qy, dz = p[3*j+2] - qz;
    float d2 = dx*dx + dy*dy + dz*dz;
    if (d2 > r2) continue;
    if (cnt < 64) {
      int c = cnt++;
      hd[c] = d2; hi[c] = j;
      while (c > 0) {
        int par = (c - 1) >> 1;
        if (hd[par] < hd[c] || (hd[par] == hd[c] && hi[par] < hi[c])) {
          float td = hd[par]; hd[par] = hd[c]; hd[c] = td;
          int   ti = hi[par]; hi[par] = hi[c]; hi[c] = ti;
          c = par;
        } else break;
      }
    } else if (d2 < hd[0] || (d2 == hd[0] && j < hi[0])) {
      hd[0] = d2; hi[0] = j;
      int c = 0;
      for (;;) {
        int l = 2*c + 1, r = l + 1, big = c;
        if (l < 64 && (hd[l] > hd[big] || (hd[l] == hd[big] && hi[l] > hi[big]))) big = l;
        if (r < 64 && (hd[r] > hd[big] || (hd[r] == hd[big] && hi[r] > hi[big]))) big = r;
        if (big == c) break;
        float td = hd[big]; hd[big] = hd[c]; hd[c] = td;
        int   ti = hi[big]; hi[big] = hi[c]; hi[c] = ti;
        c = big;
      }
    }
  }
  int* outp = nidx + (size_t)t * 64;
  for (int s = 0; s < 64; ++s) outp[s] = (s < cnt) ? hi[s] : -1;
}

// ---------------------------------------------------------------------------
// Edge features: concat(x[nidx], pos[nidx]-qpos), f16, zero-padded to Kpad.
// x == nullptr means x := pos (SA1 where features are the coordinates).
// ---------------------------------------------------------------------------
__global__ void edge_feat_kernel(const float* __restrict__ pos,     // [N,3] one frame
                                 const float* __restrict__ qpos,    // [M,3]
                                 const _Float16* __restrict__ x,    // [N,Cx] or null
                                 const int* __restrict__ nidx,      // [M*64]
                                 _Float16* __restrict__ E,          // [M*64, Kpad]
                                 int Cx, int Kpad, int total) {
  int t = blockIdx.x * 256 + threadIdx.x;
  if (t >= total) return;
  int q = t >> 6;
  int src = nidx[t];
  _Float16* e = E + (size_t)t * Kpad;
  if (src < 0) {
    for (int c = 0; c < Kpad; ++c) e[c] = (_Float16)0.0f;
    return;
  }
  const float* ps = pos + (size_t)src * 3;
  if (x) {
    const _Float16* xs = x + (size_t)src * Cx;
    for (int c = 0; c < Cx; ++c) e[c] = xs[c];
  } else {
    e[0] = (_Float16)ps[0]; e[1] = (_Float16)ps[1]; e[2] = (_Float16)ps[2];
  }
  const float* qp = qpos + (size_t)q * 3;
  e[Cx + 0] = (_Float16)(ps[0] - qp[0]);
  e[Cx + 1] = (_Float16)(ps[1] - qp[1]);
  e[Cx + 2] = (_Float16)(ps[2] - qp[2]);
  for (int c = Cx + 3; c < Kpad; ++c) e[c] = (_Float16)0.0f;
}

// ---------------------------------------------------------------------------
// WMMA GEMM helpers: 16x64 strip per wave, software-pipelined k-loop.
// ---------------------------------------------------------------------------
__device__ __forceinline__ void load_frag(const _Float16* __restrict__ ap,
                                          const _Float16* __restrict__ bp0,
                                          int Kp, int kb,
                                          v8h& alo, v8h& ahi,
                                          v8h blo[4], v8h bhi[4]) {
  alo = *(const v8h*)(ap + kb);
  ahi = *(const v8h*)(ap + kb + 16);
#pragma unroll
  for (int j = 0; j < 4; ++j) {
    const _Float16* bp = bp0 + (size_t)(j * 16) * Kp;
    blo[j] = *(const v8h*)(bp + kb);
    bhi[j] = *(const v8h*)(bp + kb + 16);
  }
}

__device__ __forceinline__ void wmma4(const v8h& alo, const v8h& ahi,
                                      const v8h blo[4], const v8h bhi[4],
                                      v8f acc[4]) {
  v16h a;
#pragma unroll
  for (int i = 0; i < 8; ++i) { a[i] = alo[i]; a[i + 8] = ahi[i]; }
#pragma unroll
  for (int j = 0; j < 4; ++j) {
    v16h b;
#pragma unroll
    for (int i = 0; i < 8; ++i) { b[i] = blo[j][i]; b[i + 8] = bhi[j][i]; }
    acc[j] = __builtin_amdgcn_wmma_f32_16x16x32_f16(false, a, false, b,
                                                    (short)0, acc[j], false, false);
  }
}

// WMMA GEMM: C[M,N] = A[M,Kp] * Wt[N,Kp]^T + bias (optional ReLU), f16 in,
// f32 accumulate, f16 out. One wave computes a 16x64 strip (4 WMMA tiles),
// reusing one A fragment across 4 v_wmma per k-step (4x less A traffic; A is
// the HBM-side stream, weights live in L2). Rotated software pipeline: the
// next k-block's 10 b128 loads are issued before the current block's WMMAs,
// so waits can be partial and load latency overlaps matrix issue. All
// per-wave guards are wave-uniform so EXEC is all-ones at every v_wmma.
// A lane layout (16-bit A 16x32): halfs 0-7 -> K=koff..koff+7,
// halfs 8-15 -> K=16+koff.., koff = (lane<16)?0:8, M = lane&15.
// B lane layout: same K order as A (halfs 0-7 -> K=boff.., 8-15 -> K=16+boff..), boff = (lane<16)?0:8, N=lane&15.
// C/D layout: acc[v] -> M = v + (lane<16?0:8), N = lane&15 (ISA table).
// Requires N % 64 == 0 (true for all WMMA layers here).
__global__ void wmma_gemm_kernel(const _Float16* __restrict__ A,   // [M, Kp]
                                 const _Float16* __restrict__ Wt,  // [N, Kp]
                                 const float* __restrict__ bias,   // [N]
                                 _Float16* __restrict__ C,         // [M, N]
                                 int M, int N, int Kp, int relu) {
  const int lane = threadIdx.x & 31;
  const int wave = threadIdx.x >> 5;
  const int mtiles  = M >> 4;
  const int ngroups = N >> 6;                 // 64-column strips
  int t = blockIdx.x * 8 + wave;
  if (t >= mtiles * ngroups) return;          // wave-uniform exit
  const int mt = t % mtiles;
  const int ng = t / mtiles;
  const int l15  = lane & 15;
  const int koff = (lane < 16) ? 0 : 8;
  const int boff = (lane < 16) ? 0 : 8;
  const _Float16* ap  = A  + (size_t)(mt * 16 + l15) * Kp + koff;
  const _Float16* bp0 = Wt + (size_t)(ng * 64 + l15) * Kp + boff;
  v8f acc[4] = {};
  v8h alo, ahi, blo[4], bhi[4];
  load_frag(ap, bp0, Kp, 0, alo, ahi, blo, bhi);
  for (int kb = 32; kb < Kp; kb += 32) {
    v8h nalo, nahi, nblo[4], nbhi[4];
    load_frag(ap, bp0, Kp, kb, nalo, nahi, nblo, nbhi); // prefetch next block
    wmma4(alo, ahi, blo, bhi, acc);                     // consume current
    alo = nalo; ahi = nahi;
#pragma unroll
    for (int j = 0; j < 4; ++j) { blo[j] = nblo[j]; bhi[j] = nbhi[j]; }
  }
  wmma4(alo, ahi, blo, bhi, acc);                       // drain
  const int mbase = mt * 16 + ((lane < 16) ? 0 : 8);
#pragma unroll
  for (int j = 0; j < 4; ++j) {
    const int n = ng * 64 + j * 16 + l15;
    const float bs = bias[n];
#pragma unroll
    for (int v = 0; v < 8; ++v) {
      float r = acc[j][v] + bs;
      if (relu) r = fmaxf(r, 0.0f);
      C[(size_t)(mbase + v) * N + n] = (_Float16)r;
    }
  }
}

// Max aggregation over K=64 neighbor slots with validity mask; no valid -> 0.
__global__ void maxagg_kernel(const _Float16* __restrict__ msg, // [M*64, C]
                              const int* __restrict__ nidx,     // [M*64]
                              _Float16* __restrict__ out,       // [M, C]
                              int C, int total) {
  int t = blockIdx.x * 256 + threadIdx.x;
  if (t >= total) return;
  int q = t / C, c = t - q * C;
  const _Float16* m = msg + (size_t)q * 64 * C + c;
  const int* ni = nidx + (size_t)q * 64;
  float best = -1.0e30f; bool any = false;
  for (int k = 0; k < 64; ++k) {
    if (ni[k] >= 0) {
      float v = (float)m[(size_t)k * C];
      if (!any || v > best) best = v;
      any = true;
    }
  }
  out[t] = (_Float16)(any ? best : 0.0f);
}

__global__ void concat_sa3_kernel(const _Float16* __restrict__ x2,  // [R,256]
                                  const float* __restrict__ qpos2,  // [R,3]
                                  _Float16* __restrict__ A,         // [R,288]
                                  int total) {
  int t = blockIdx.x * 256 + threadIdx.x;
  if (t >= total) return;
  int row = t / 288, c = t - row * 288;
  _Float16 v;
  if (c < 256)      v = x2[(size_t)row * 256 + c];
  else if (c < 259) v = (_Float16)qpos2[(size_t)row * 3 + (c - 256)];
  else              v = (_Float16)0.0f;
  A[t] = v;
}

__global__ void framemax_kernel(const _Float16* __restrict__ g, // [F*256,1024]
                                _Float16* __restrict__ feats,   // [F,1024]
                                int total) {
  int t = blockIdx.x * 256 + threadIdx.x;
  if (t >= total) return;
  int f = t / 1024, c = t - f * 1024;
  const _Float16* p = g + (size_t)f * 256 * 1024 + c;
  float best = (float)p[0];
  for (int r = 1; r < 256; ++r) {
    float v = (float)p[(size_t)r * 1024];
    if (v > best) best = v;
  }
  feats[t] = (_Float16)best;
}

// Weight prep: W[K,N] f32 -> Wt[N,Kp] f16 (transposed, zero K-padding).
__global__ void wconv_kernel(const float* __restrict__ W, _Float16* __restrict__ Wt,
                             int Kd, int Nd, int Kp, int total) {
  int t = blockIdx.x * 256 + threadIdx.x;
  if (t >= total) return;
  int n = t / Kp, k = t - n * Kp;
  Wt[t] = (k < Kd) ? (_Float16)W[(size_t)k * Nd + n] : (_Float16)0.0f;
}

// Final classifier layer (N=6 not a WMMA tile): [16,256] x [256,6] + b -> f32.
__global__ void final_linear_kernel(const _Float16* __restrict__ act,
                                    const float* __restrict__ W,
                                    const float* __restrict__ b,
                                    float* __restrict__ out) {
  int t = blockIdx.x * 96 + threadIdx.x;
  if (t >= 96) return;
  int f = t / 6, o = t - f * 6;
  float acc = b[o];
  const _Float16* a = act + (size_t)f * 256;
  for (int k = 0; k < 256; ++k) acc += (float)a[k] * W[(size_t)k * 6 + o];
  out[t] = acc;
}

extern "C" void kernel_launch(void* const* d_in, const int* in_sizes, int n_in,
                              void* d_out, int out_size, void* d_ws, size_t ws_size,
                              hipStream_t stream) {
  (void)in_sizes; (void)n_in; (void)out_size; (void)ws_size;
  const float* data = (const float*)d_in[0];
  const float* w[12]; const float* b[12];
  for (int i = 0; i < 12; ++i) {
    w[i] = (const float*)d_in[1 + 2 * i];
    b[i] = (const float*)d_in[2 + 2 * i];
  }
  struct L { int K, N, Kp; };
  const L layers[12] = {
    {6, 64, 32}, {64, 64, 64}, {64, 128, 64},          // sa1
    {131, 128, 160}, {128, 128, 128}, {128, 256, 128}, // sa2
    {259, 256, 288}, {256, 512, 256}, {512, 1024, 512},// sa3
    {1024, 512, 1024}, {512, 256, 512}, {256, 6, 0}    // cls
  };

  // ---- workspace carve (256B aligned) ----
  char* p = (char*)d_ws;
  auto alloc = [&](size_t bytes) -> char* {
    char* r = p; p += (bytes + 255) & ~(size_t)255; return r;
  };
  _Float16* wt[11];
  for (int i = 0; i < 11; ++i)
    wt[i] = (_Float16*)alloc((size_t)layers[i].N * layers[i].Kp * 2);
  float* qpos1 = (float*)alloc((size_t)16 * 1024 * 3 * 4);
  float* qpos2 = (float*)alloc((size_t)16 * 256 * 3 * 4);
  int* idx1  = (int*)alloc((size_t)16 * 1024 * 4);
  int* idx2  = (int*)alloc((size_t)16 * 256 * 4);
  int* nidx1 = (int*)alloc((size_t)16 * 1024 * 64 * 4);
  int* nidx2 = (int*)alloc((size_t)16 * 256 * 64 * 4);
  _Float16* x1    = (_Float16*)alloc((size_t)16 * 1024 * 128 * 2);
  _Float16* x2    = (_Float16*)alloc((size_t)16 * 256 * 256 * 2);
  _Float16* feats = (_Float16*)alloc((size_t)16 * 1024 * 2);
  _Float16* bufA  = (_Float16*)alloc((size_t)16 << 20);
  _Float16* bufB  = (_Float16*)alloc((size_t)16 << 20);

  auto gemm = [&](const _Float16* A, int li, _Float16* C, int M, int relu) {
    int tiles = (M >> 4) * (layers[li].N >> 6);   // 16x64 strips per wave
    wmma_gemm_kernel<<<(tiles + 7) / 8, 256, 0, stream>>>(
        A, wt[li], b[li], C, M, layers[li].N, layers[li].Kp, relu);
  };

  // ---- weight conversion ----
  for (int i = 0; i < 11; ++i) {
    int tot = layers[i].N * layers[i].Kp;
    wconv_kernel<<<(tot + 255) / 256, 256, 0, stream>>>(
        w[i], wt[i], layers[i].K, layers[i].N, layers[i].Kp, tot);
  }

  // ---- SA1: fps(2048->1024), knn r=0.2, edge MLP 6->64->64->128, max ----
  fps_kernel<<<16, 256, 0, stream>>>(data, idx1, 2048, 1024);
  gather_pos_kernel<<<(16 * 1024 + 255) / 256, 256, 0, stream>>>(
      data, idx1, qpos1, 2048, 1024, 16 * 1024);
  knn_kernel<<<(16 * 1024 + 255) / 256, 256, 0, stream>>>(
      data, qpos1, nidx1, 2048, 1024, 16 * 1024, 0.2f * 0.2f);
  for (int f = 0; f < 16; ++f) {
    const float* posf = data  + (size_t)f * 2048 * 3;
    const float* qpf  = qpos1 + (size_t)f * 1024 * 3;
    const int*   nif  = nidx1 + (size_t)f * 1024 * 64;
    const int ne = 1024 * 64;
    edge_feat_kernel<<<(ne + 255) / 256, 256, 0, stream>>>(
        posf, qpf, (const _Float16*)nullptr, nif, bufA, 3, 32, ne);
    gemm(bufA, 0, bufB, ne, 1);
    gemm(bufB, 1, bufA, ne, 1);
    gemm(bufA, 2, bufB, ne, 0);
    maxagg_kernel<<<(1024 * 128 + 255) / 256, 256, 0, stream>>>(
        bufB, nif, x1 + (size_t)f * 1024 * 128, 128, 1024 * 128);
  }

  // ---- SA2: fps(1024->256), knn r=0.4, edge MLP 131->128->128->256, max ----
  fps_kernel<<<16, 256, 0, stream>>>(qpos1, idx2, 1024, 256);
  gather_pos_kernel<<<(16 * 256 + 255) / 256, 256, 0, stream>>>(
      qpos1, idx2, qpos2, 1024, 256, 16 * 256);
  knn_kernel<<<(16 * 256 + 255) / 256, 256, 0, stream>>>(
      qpos1, qpos2, nidx2, 1024, 256, 16 * 256, 0.4f * 0.4f);
  for (int f = 0; f < 16; ++f) {
    const float*    posf = qpos1 + (size_t)f * 1024 * 3;
    const float*    qpf  = qpos2 + (size_t)f * 256 * 3;
    const int*      nif  = nidx2 + (size_t)f * 256 * 64;
    const _Float16* xf   = x1 + (size_t)f * 1024 * 128;
    const int ne = 256 * 64;
    edge_feat_kernel<<<(ne + 255) / 256, 256, 0, stream>>>(
        posf, qpf, xf, nif, bufA, 128, 160, ne);
    gemm(bufA, 3, bufB, ne, 1);
    gemm(bufB, 4, bufA, ne, 1);
    gemm(bufA, 5, bufB, ne, 0);
    maxagg_kernel<<<(256 * 256 + 255) / 256, 256, 0, stream>>>(
        bufB, nif, x2 + (size_t)f * 256 * 256, 256, 256 * 256);
  }

  // ---- SA3 (global): concat(x2,p2) MLP 259->256->512->1024, frame max ----
  const int M3 = 16 * 256;
  concat_sa3_kernel<<<(M3 * 288 + 255) / 256, 256, 0, stream>>>(
      x2, qpos2, bufA, M3 * 288);
  gemm(bufA, 6, bufB, M3, 1);
  gemm(bufB, 7, bufA, M3, 1);
  gemm(bufA, 8, bufB, M3, 0);
  framemax_kernel<<<(16 * 1024 + 255) / 256, 256, 0, stream>>>(
      bufB, feats, 16 * 1024);

  // ---- classifier: 1024->512->256->6 ----
  gemm(feats, 9, bufA, 16, 1);
  gemm(bufA, 10, bufB, 16, 1);
  final_linear_kernel<<<1, 96, 0, stream>>>(bufB, w[11], b[11], (float*)d_out);
}
